// PatchMoELayer_37915971289528
// MI455X (gfx1250) — matrix operand order repaired and checked by reading, not compiled, not run on hardware
//
#include <hip/hip_runtime.h>
#include <hip/hip_bf16.h>

typedef __attribute__((ext_vector_type(16))) __bf16 bf16x16;
typedef __attribute__((ext_vector_type(8)))  float  f32x8;

#define C_DIM 256
#define FF_DIM 1024
#define E_DIM 4
#define NPATCH 4096          // B * P
#define MPB 4                // patches per GEMM block (M = 64)
#define STRA 264             // LDS A row stride (bf16): 256 + 8 pad  -> conflict-free b128 reads
#define STRH 1048            // LDS h row stride (bf16): 1024 + 24 pad -> conflict-free b128 reads
#define LDS_BYTES (64 * STRA * 2 + 64 * STRH * 2)   // 167,936 B (dynamic LDS; < 320KB/WGP)

union FragU { uint4 u[2]; bf16x16 v; };

static __device__ __forceinline__ unsigned short f2bf(float f) {
    unsigned int u = __float_as_uint(f);
    unsigned int r = (u + 0x7FFFu + ((u >> 16) & 1u)) >> 16;   // RNE
    return (unsigned short)r;
}

static __device__ __forceinline__ float fast_tanh(float x) {
#if __has_builtin(__builtin_amdgcn_tanhf)
    return __builtin_amdgcn_tanhf(x);                 // CDNA5 TRANS op
#elif __has_builtin(__builtin_amdgcn_tanh_f32)
    return __builtin_amdgcn_tanh_f32(x);
#else
    float e = __expf(2.0f * x);                       // branch-free fallback (v_exp_f32)
    return 1.0f - 2.0f / (e + 1.0f);
#endif
}

static __device__ __forceinline__ float gelu_f(float v) {
    // tanh-form GELU (branch-free; avoids libm erf's divergent polynomial path)
    float t = fast_tanh(0.7978845608028654f * (v + 0.044715f * v * v * v));
    return 0.5f * v * (1.0f + t);
}

// ---------------- zero out + counts ----------------
__global__ __launch_bounds__(256) void zero_kernel(float4* __restrict__ out4, int n4,
                                                   int* __restrict__ counts) {
    int i = blockIdx.x * 256 + threadIdx.x;
    if (i < n4) out4[i] = make_float4(0.f, 0.f, 0.f, 0.f);
    if (blockIdx.x == 0 && threadIdx.x < E_DIM) counts[threadIdx.x] = 0;
}

// ---------------- fp32 -> bf16 weight conversion (4 elems/thread) ----------------
__global__ __launch_bounds__(256) void cvt_bf16_kernel(const float* __restrict__ src,
                                                       unsigned short* __restrict__ dst, int n) {
    int i = (blockIdx.x * 256 + threadIdx.x) * 4;
    if (i + 3 < n) {
        float4 f = *(const float4*)(src + i);
        ushort4 o;
        o.x = f2bf(f.x); o.y = f2bf(f.y); o.z = f2bf(f.z); o.w = f2bf(f.w);
        *(ushort4*)(dst + i) = o;
    }
}

// ---------------- router: patch mean -> gate -> softmax -> top2 -> expert lists ----------------
__global__ __launch_bounds__(256) void router_kernel(const float* __restrict__ x,
                                                     const float* __restrict__ Wg,
                                                     int* __restrict__ counts,
                                                     int2* __restrict__ assign) {
    __shared__ float smean[C_DIM];
    __shared__ float slog[E_DIM];
    int p   = blockIdx.x;           // 0..4095
    int b   = p >> 10;
    int pid = p & 1023;
    int nhi = pid >> 5, nwi = pid & 31;
    int c = threadIdx.x;

    float s = 0.0f;
    #pragma unroll
    for (int t = 0; t < 16; ++t) {
        int h = nhi * 4 + (t >> 2);
        int w = nwi * 4 + (t & 3);
        s += x[(((b * 128 + h) * 128) + w) * C_DIM + c];
    }
    smean[c] = s * 0.0625f;
    __syncthreads();

    if (c < E_DIM) {
        float acc = 0.0f;
        #pragma unroll 8
        for (int k = 0; k < C_DIM; ++k) acc += smean[k] * Wg[c * C_DIM + k];
        slog[c] = acc;
    }
    __syncthreads();

    if (c == 0) {
        float m = slog[0];
        #pragma unroll
        for (int e = 1; e < E_DIM; ++e) m = fmaxf(m, slog[e]);
        float pe[E_DIM]; float S = 0.0f;
        #pragma unroll
        for (int e = 0; e < E_DIM; ++e) { pe[e] = __expf(slog[e] - m); S += pe[e]; }
        int i0 = 0;
        #pragma unroll
        for (int e = 1; e < E_DIM; ++e) if (pe[e] > pe[i0]) i0 = e;
        int i1 = (i0 == 0) ? 1 : 0;
        #pragma unroll
        for (int e = 0; e < E_DIM; ++e) if (e != i0 && pe[e] > pe[i1]) i1 = e;
        float w0 = pe[i0] / S, w1 = pe[i1] / S;
        float inv = 1.0f / (w0 + w1 + 1e-9f);
        int pos0 = atomicAdd(&counts[i0], 1);
        assign[i0 * NPATCH + pos0] = make_int2(p, __float_as_int(w0 * inv));
        int pos1 = atomicAdd(&counts[i1], 1);
        assign[i1 * NPATCH + pos1] = make_int2(p, __float_as_int(w1 * inv));
    }
}

// ---------------- grouped expert FFN: 4 patches (M=64) x 1 expert per block ----------------
// 512 threads = 16 waves. Each B-fragment feeds 4 WMMAs (one per patch tile) -> 4x less L2 weight traffic.
__global__ __launch_bounds__(512) void moe_gemm_kernel(const float* __restrict__ x,
                                                       const float* __restrict__ b1,
                                                       const float* __restrict__ b2,
                                                       const unsigned short* __restrict__ w1bf,
                                                       const unsigned short* __restrict__ w2bf,
                                                       const int* __restrict__ counts,
                                                       const int2* __restrict__ assign,
                                                       float* __restrict__ out) {
    extern __shared__ char smem[];
    unsigned short* sA = (unsigned short*)smem;                       // 64 x 256 (+pad) bf16 tokens
    unsigned short* sH = (unsigned short*)(smem + 64 * STRA * 2);     // 64 x 1024 (+pad) bf16 hidden
    __shared__ int   sPid[MPB];
    __shared__ float sWgt[MPB];

    int e    = blockIdx.x >> 10;          // expert
    int grp  = blockIdx.x & 1023;         // group within expert
    int base = grp * MPB;
    int cnt  = counts[e];
    if (base >= cnt) return;              // uniform early exit (static grid, data-dep work)
    int nP = cnt - base; if (nP > MPB) nP = MPB;

    int tid = threadIdx.x;
    if (tid < MPB) {
        int idx = tid < nP ? tid : nP - 1;
        int2 a = assign[e * NPATCH + base + idx];
        sPid[tid] = a.x;
        sWgt[tid] = __int_as_float(a.y);
    }
    __syncthreads();

    // Phase 0: load 64 tokens x 256 ch fp32, convert -> bf16 into sA
    {
        int row = tid >> 3;               // 0..63 (slot*16 + token)
        int cb  = (tid & 7) * 32;         // 32 channels per thread
        int s   = row >> 4;
        int t   = row & 15;
        int p   = sPid[s];
        int b   = p >> 10;
        int pid = p & 1023;
        int h = (pid >> 5) * 4 + (t >> 2);
        int w = (pid & 31) * 4 + (t & 3);
        const float* src = x + (((b * 128 + h) * 128) + w) * C_DIM + cb;
        unsigned short* dst = sA + row * STRA + cb;
        #pragma unroll
        for (int i = 0; i < 32; i += 4) {
            float4 f = *(const float4*)(src + i);
            dst[i + 0] = f2bf(f.x); dst[i + 1] = f2bf(f.y);
            dst[i + 2] = f2bf(f.z); dst[i + 3] = f2bf(f.w);
        }
    }
    __syncthreads();

    int wave = tid >> 5;
    int lane = tid & 31;
    int hl = lane >> 4;       // half-wave select
    int ln = lane & 15;       // row (A) / column (B,C,D)

    const unsigned short* W1e = w1bf + e * (FF_DIM * C_DIM);
    const unsigned short* W2e = w2bf + e * (C_DIM * FF_DIM);

    // ---- GEMM1: [64 x 256] @ [256 x 1024] -> h ; 4 N-tiles per wave, B shared across 4 M-tiles ----
    for (int i = 0; i < 4; ++i) {
        int nt = wave * 4 + i;
        f32x8 acc[MPB];
        #pragma unroll
        for (int mt = 0; mt < MPB; ++mt) acc[mt] = (f32x8){};
        #pragma unroll
        for (int kt = 0; kt < 8; ++kt) {
            const unsigned short* bp = W1e + (nt * 16 + ln) * C_DIM + kt * 32 + hl * 16;
            FragU bf;
            bf.u[0] = *(const uint4*)(bp);
            bf.u[1] = *(const uint4*)(bp + 8);
            #pragma unroll
            for (int mt = 0; mt < MPB; ++mt) {
                const unsigned short* ap = sA + (mt * 16 + ln) * STRA + kt * 32 + hl * 16;
                FragU af;
                af.u[0] = *(const uint4*)(ap);
                af.u[1] = *(const uint4*)(ap + 8);
                acc[mt] = __builtin_amdgcn_wmma_f32_16x16x32_bf16(
                    false, af.v, false, bf.v, (short)0, acc[mt], false, false);
            }
        }
        float b1v = b1[e * FF_DIM + nt * 16 + ln];
        #pragma unroll
        for (int mt = 0; mt < MPB; ++mt) {
            #pragma unroll
            for (int r = 0; r < 8; ++r) {
                float v = gelu_f(acc[mt][r] + b1v);
                sH[(mt * 16 + hl * 8 + r) * STRH + nt * 16 + ln] = f2bf(v);
            }
        }
    }
    __syncthreads();

    // ---- GEMM2: [64 x 1024] @ [1024 x 256] -> y ; 1 N-tile per wave, B shared across 4 M-tiles ----
    {
        int nt = wave;                    // 0..15 covers C=256
        f32x8 acc[MPB];
        #pragma unroll
        for (int mt = 0; mt < MPB; ++mt) acc[mt] = (f32x8){};
        #pragma unroll 4
        for (int kt = 0; kt < 32; ++kt) {
            const unsigned short* bp = W2e + (nt * 16 + ln) * FF_DIM + kt * 32 + hl * 16;
            FragU bf;
            bf.u[0] = *(const uint4*)(bp);
            bf.u[1] = *(const uint4*)(bp + 8);
            #pragma unroll
            for (int mt = 0; mt < MPB; ++mt) {
                const unsigned short* ap = sH + (mt * 16 + ln) * STRH + kt * 32 + hl * 16;
                FragU af;
                af.u[0] = *(const uint4*)(ap);
                af.u[1] = *(const uint4*)(ap + 8);
                acc[mt] = __builtin_amdgcn_wmma_f32_16x16x32_bf16(
                    false, af.v, false, bf.v, (short)0, acc[mt], false, false);
            }
        }
        float b2v = b2[e * C_DIM + nt * 16 + ln];
        int c = nt * 16 + ln;
        for (int mt = 0; mt < MPB; ++mt) {
            if (mt >= nP) break;          // uniform across block
            float wgt = sWgt[mt];
            int p   = sPid[mt];
            int b   = p >> 10;
            int pid = p & 1023;
            int hb = (pid >> 5) * 4;
            int wb = (pid & 31) * 4;
            #pragma unroll
            for (int r = 0; r < 8; ++r) {
                int t = hl * 8 + r;
                int h = hb + (t >> 2);
                int w = wb + (t & 3);
                float val = wgt * (acc[mt][r] + b2v);
                // exactly 2 contributions per element; fp add is commutative -> deterministic
                __hip_atomic_fetch_add(&out[(((b * 128 + h) * 128) + w) * C_DIM + c],
                                       val, __ATOMIC_RELAXED, __HIP_MEMORY_SCOPE_AGENT);
            }
        }
    }
}

extern "C" void kernel_launch(void* const* d_in, const int* in_sizes, int n_in,
                              void* d_out, int out_size, void* d_ws, size_t ws_size,
                              hipStream_t stream) {
    const float* x  = (const float*)d_in[0];   // [4,128,128,256]
    const float* Wg = (const float*)d_in[1];   // [4,256]
    const float* W1 = (const float*)d_in[2];   // [4,1024,256]
    const float* b1 = (const float*)d_in[3];   // [4,1024]
    const float* W2 = (const float*)d_in[4];   // [4,256,1024]
    const float* b2 = (const float*)d_in[5];   // [4,256]
    float* out = (float*)d_out;

    // ws: [0,2MB) W1 bf16 | [2MB,4MB) W2 bf16 | +counts int[4] | +assign int2[4][4096]
    char* ws = (char*)d_ws;
    unsigned short* w1bf  = (unsigned short*)(ws);
    unsigned short* w2bf  = (unsigned short*)(ws + 2097152);
    int*            counts = (int*)(ws + 4194304);
    int2*           assign = (int2*)(ws + 4194368);

    const int n4 = out_size / 4;               // 4,194,304 float4
    zero_kernel<<<(n4 + 255) / 256, 256, 0, stream>>>((float4*)out, n4, counts);

    const int nW = E_DIM * FF_DIM * C_DIM;     // 1,048,576 elements each
    cvt_bf16_kernel<<<nW / 1024, 256, 0, stream>>>(W1, w1bf, nW);
    cvt_bf16_kernel<<<nW / 1024, 256, 0, stream>>>(W2, w2bf, nW);

    router_kernel<<<NPATCH, 256, 0, stream>>>(x, Wg, counts, assign);

    // static grid: 4 experts x 1024 groups; blocks beyond per-expert count exit uniformly
    moe_gemm_kernel<<<E_DIM * 1024, 512, LDS_BYTES, stream>>>(x, b1, b2, w1bf, w2bf,
                                                              counts, assign, out);
}
